// PointNet2ClsMsg_86363202388415
// MI455X (gfx1250) — matrix operand order repaired and checked by reading, not compiled, not run on hardware
//
#include <hip/hip_runtime.h>
#include <cstddef>

// PointNet++ MSG classifier, fused for MI455X (gfx1250, wave32, WMMA).
// - Per-center fused 3-layer MLP + K-maxpool entirely in LDS (chunked over K),
//   using v_wmma_f32_16x16x32_f16 with f16 activations/weights, f32 accumulate.
// - FPS: one block/batch, dist array in LDS, shuffle argmax reduction.
// - Ball query: one wave/center, ballot-based ordered compaction.

typedef _Float16 f16;
typedef __attribute__((ext_vector_type(16))) _Float16 v16h;
typedef __attribute__((ext_vector_type(8)))  float    v8f;

#define BN_SCALE 0.9999950000374997f

// ---------------------------------------------------------------------------
// Weight conversion: f32 [cout][cin] -> f16 [cout][cinp] zero-padded
// ---------------------------------------------------------------------------
__global__ void convw_kernel(const float* __restrict__ src, f16* __restrict__ dst,
                             int cout, int cin, int cinp) {
    int t = blockIdx.x * blockDim.x + threadIdx.x;
    int tot = cout * cinp;
    if (t >= tot) return;
    int o = t / cinp, ch = t - o * cinp;
    dst[t] = (ch < cin) ? (f16)src[o * cin + ch] : (f16)0.0f;
}

// ---------------------------------------------------------------------------
// Farthest point sampling: grid = B, block = 256. Writes indices + gathered xyz.
// ---------------------------------------------------------------------------
__global__ void fps_kernel(const float* __restrict__ xyz, int N, int npoint,
                           int* __restrict__ outidx, float* __restrict__ new_xyz) {
    const int b   = blockIdx.x;
    const int tid = threadIdx.x;
    __shared__ float sdist[2048];
    __shared__ float wmax[8];
    __shared__ int   widx[8];
    __shared__ int   sfar;
    const float* bx = xyz + (size_t)b * N * 3;
    for (int i = tid; i < N; i += blockDim.x) sdist[i] = 1e10f;
    __syncthreads();
    int far = 0;
    for (int it = 0; it < npoint; ++it) {
        if (tid == 0) outidx[b * npoint + it] = far;
        if (tid < 3)  new_xyz[((size_t)b * npoint + it) * 3 + tid] = bx[far * 3 + tid];
        const float cx = bx[far * 3 + 0], cy = bx[far * 3 + 1], cz = bx[far * 3 + 2];
        float best = -1.0f; int bidx = 0;
        for (int i = tid; i < N; i += blockDim.x) {
            float dx = bx[i * 3 + 0] - cx;
            float dy = bx[i * 3 + 1] - cy;
            float dz = bx[i * 3 + 2] - cz;
            float d  = dx * dx + dy * dy + dz * dz;
            float nd = fminf(sdist[i], d);
            sdist[i] = nd;
            if (nd > best) { best = nd; bidx = i; }
        }
        // wave32 reduction (max dist, tie -> lower index)
        for (int off = 16; off > 0; off >>= 1) {
            float ob = __shfl_xor(best, off);
            int   oi = __shfl_xor(bidx, off);
            if (ob > best || (ob == best && oi < bidx)) { best = ob; bidx = oi; }
        }
        if ((tid & 31) == 0) { wmax[tid >> 5] = best; widx[tid >> 5] = bidx; }
        __syncthreads();
        if (tid == 0) {
            float bb = wmax[0]; int bi = widx[0];
            for (int w = 1; w < (int)(blockDim.x >> 5); ++w)
                if (wmax[w] > bb || (wmax[w] == bb && widx[w] < bi)) { bb = wmax[w]; bi = widx[w]; }
            sfar = bi;
        }
        __syncthreads();
        far = sfar;
    }
}

// ---------------------------------------------------------------------------
// Ball query: one wave per center; first K in-ball indices in index order,
// pad misses with first hit.
// ---------------------------------------------------------------------------
__global__ void ballquery_kernel(const float* __restrict__ xyz, int N,
                                 const float* __restrict__ new_xyz,
                                 int total, int S, float r2, int K,
                                 int* __restrict__ idx) {
    const int gw   = (blockIdx.x * blockDim.x + threadIdx.x) >> 5;
    const int lane = threadIdx.x & 31;
    if (gw >= total) return;
    const int b = gw / S;
    const float cx = new_xyz[(size_t)gw * 3 + 0];
    const float cy = new_xyz[(size_t)gw * 3 + 1];
    const float cz = new_xyz[(size_t)gw * 3 + 2];
    const float* bx = xyz + (size_t)b * N * 3;
    int cnt = 0, first = -1;
    for (int base = 0; base < N; base += 32) {
        int i = base + lane;
        bool in = false;
        if (i < N) {
            float dx = bx[i * 3 + 0] - cx;
            float dy = bx[i * 3 + 1] - cy;
            float dz = bx[i * 3 + 2] - cz;
            in = (dx * dx + dy * dy + dz * dz) <= r2;
        }
        unsigned long long m = __ballot(in) & 0xffffffffull;
        if (in) {
            int pos = cnt + (int)__popcll(m & ((1ull << lane) - 1ull));
            if (pos < K) idx[(size_t)gw * K + pos] = i;
        }
        int c = (int)__popcll(m);
        if (first < 0 && c > 0) first = base + (__ffsll(m) - 1);
        cnt += c;
        if (cnt >= K) break;
    }
    if (cnt > K) cnt = K;
    int f = (first < 0) ? 0 : first;
    for (int p = cnt + lane; p < K; p += 32) idx[(size_t)gw * K + p] = f;
}

// ---------------------------------------------------------------------------
// One WMMA GEMM layer: Dst[rows][cout] = relu(BN_SCALE*(A[rows][kdim] @ W^T + b))
// A in LDS (f16, leading dim lda); W f16 global [cout][ldw]; Dst LDS f16.
// rows in {16,32}; kdim % 32 == 0; cout % 16 == 0. 8 waves per block.
// ---------------------------------------------------------------------------
__device__ __forceinline__ void mlp_layer_wmma(
    const f16* __restrict__ A, int lda, f16* __restrict__ Dst, int ldd,
    const f16* __restrict__ W, int ldw, const float* __restrict__ bias,
    int rows, int kdim, int cout, int wave, int lane) {
    const int g = lane >> 4;      // lane half
    const int r = lane & 15;
    const int rowt_cnt = rows >> 4;
    const int rowt  = wave % rowt_cnt;
    const int wstep = 8 / rowt_cnt;
    for (int ct = wave / rowt_cnt; ct * 16 < cout; ct += wstep) {
        const int n0 = ct * 16;
        v8f c;
        const float bv = bias[n0 + r];     // C/D layout: N = lane%16
#pragma unroll
        for (int j = 0; j < 8; ++j) c[j] = bv;
        const f16* arow = A + (rowt * 16 + r) * lda;   // A layout: M = lane%16
        const f16* wrow = W + (size_t)(n0 + r) * ldw;  // B layout: N = lane%16
        for (int k0 = 0; k0 < kdim; k0 += 32) {
            v16h af, bf;
            const f16* ap = arow + k0 + g * 8;         // K = k0+g*8.. and +16
#pragma unroll
            for (int h = 0; h < 8; ++h) { af[h] = ap[h]; af[h + 8] = ap[h + 16]; }
            const f16* bp = wrow + k0 + g * 8;         // K = k0+g*8.. and +16
#pragma unroll
            for (int h = 0; h < 8; ++h) { bf[h] = bp[h]; bf[h + 8] = bp[h + 16]; }
            c = __builtin_amdgcn_wmma_f32_16x16x32_f16(
                    false, af, false, bf, (short)0, c, false, false);
        }
#pragma unroll
        for (int j = 0; j < 8; ++j) {      // D element: M = j + 8*g, N = r
            float v = c[j] * BN_SCALE;
            v = v > 0.0f ? v : 0.0f;
            Dst[(rowt * 16 + g * 8 + j) * ldd + (n0 + r)] = (f16)v;
        }
    }
}

// ---------------------------------------------------------------------------
// Fused set-abstraction branch: gather group -> 3x (conv+BN+ReLU) -> max over K.
// MODE 0: SA1 (centered xyz only). MODE 1: SA2 (points then centered xyz).
// MODE 2: SA3 group_all (xyz then points, no centering, idx = identity).
// Grid: one block (256 thr) per center.
// ---------------------------------------------------------------------------
template<int MODE, int KPTS, int CINP, int C0, int C1, int C2, int CHUNK, int CPTS>
__global__ __launch_bounds__(256)
void sa_mlp_kernel(const float* __restrict__ xyz, int Nin,
                   const float* __restrict__ pts,
                   const float* __restrict__ new_xyz,
                   const int* __restrict__ idx,
                   const f16* __restrict__ W0, const float* __restrict__ B0,
                   const f16* __restrict__ W1, const float* __restrict__ B1,
                   const f16* __restrict__ W2, const float* __restrict__ B2,
                   float* __restrict__ out, int S, int out_stride, int out_off) {
    constexpr int LDA = (CINP > C1 ? CINP : C1) + 8;
    constexpr int LDB = (C0 > C2 ? C0 : C2) + 8;
    __shared__ f16  bufA[CHUNK * LDA];
    __shared__ f16  bufB[CHUNK * LDB];
    __shared__ float red[C2];
    __shared__ float cen[3];
    const int bs   = blockIdx.x;
    const int b    = bs / S;
    const int tid  = threadIdx.x;
    const int wave = tid >> 5, lane = tid & 31;
    if constexpr (MODE != 2) {
        if (tid < 3) cen[tid] = new_xyz[(size_t)bs * 3 + tid];
    }
    __syncthreads();
    constexpr int NCH = KPTS / CHUNK;
    for (int ck = 0; ck < NCH; ++ck) {
        const int p0 = ck * CHUNK;
        // -------- stage grouped input chunk into bufA (f16, zero-padded) ----
        for (int e = tid; e < CHUNK * CINP; e += 256) {
            int row = e / CINP, ch = e - row * CINP;
            int p = p0 + row;
            float v = 0.0f;
            if constexpr (MODE == 0) {
                if (ch < 3) {
                    int pi = idx[(size_t)bs * KPTS + p];
                    v = xyz[((size_t)b * Nin + pi) * 3 + ch] - cen[ch];
                }
            } else if constexpr (MODE == 1) {
                int pi = idx[(size_t)bs * KPTS + p];
                if (ch < CPTS)          v = pts[((size_t)b * Nin + pi) * CPTS + ch];
                else if (ch < CPTS + 3) v = xyz[((size_t)b * Nin + pi) * 3 + (ch - CPTS)] - cen[ch - CPTS];
            } else {
                int pi = p;  // group_all
                if (ch < 3)             v = xyz[((size_t)b * Nin + pi) * 3 + ch];
                else if (ch < 3 + CPTS) v = pts[((size_t)b * Nin + pi) * CPTS + (ch - 3)];
            }
            bufA[row * LDA + ch] = (f16)v;
        }
        __syncthreads();
        mlp_layer_wmma(bufA, LDA, bufB, LDB, W0, CINP, B0, CHUNK, CINP, C0, wave, lane);
        __syncthreads();
        mlp_layer_wmma(bufB, LDB, bufA, LDA, W1, C0,   B1, CHUNK, C0,   C1, wave, lane);
        __syncthreads();
        mlp_layer_wmma(bufA, LDA, bufB, LDB, W2, C1,   B2, CHUNK, C1,   C2, wave, lane);
        __syncthreads();
        // -------- running max over this chunk's rows ------------------------
        for (int c = tid; c < C2; c += 256) {
            float m = (float)bufB[c];
            for (int rr = 1; rr < CHUNK; ++rr) m = fmaxf(m, (float)bufB[rr * LDB + c]);
            red[c] = (ck == 0) ? m : fmaxf(red[c], m);
        }
        __syncthreads();
    }
    for (int c = tid; c < C2; c += 256)
        out[(size_t)bs * out_stride + out_off + c] = red[c];
}

// ---------------------------------------------------------------------------
// FC head layer (tiny: M=8) — plain f32 dot products.
// ---------------------------------------------------------------------------
__global__ void fc_kernel(const float* __restrict__ in, int cin,
                          const float* __restrict__ W, const float* __restrict__ bias,
                          float* __restrict__ out, int cout, int nb, int relu) {
    int t = blockIdx.x * blockDim.x + threadIdx.x;
    if (t >= nb * cout) return;
    int bb = t / cout, o = t - bb * cout;
    float acc = bias[o];
    const float* ip = in + (size_t)bb * cin;
    const float* wrow = W + (size_t)o * cin;
    for (int k = 0; k < cin; ++k) acc += ip[k] * wrow[k];
    if (relu) { acc *= BN_SCALE; acc = fmaxf(acc, 0.0f); }
    out[t] = acc;
}

// ---------------------------------------------------------------------------
// Host orchestration
// ---------------------------------------------------------------------------
extern "C" void kernel_launch(void* const* d_in, const int* in_sizes, int n_in,
                              void* d_out, int out_size, void* d_ws, size_t ws_size,
                              hipStream_t stream) {
    (void)in_sizes; (void)n_in; (void)out_size; (void)ws_size;
    const int B = 8, N = 2048;
    const float* pc = (const float*)d_in[0];

    char* ws = (char*)d_ws;
    size_t off = 0;
    auto alloc = [&](size_t bytes) -> void* {
        void* p = ws + off;
        off += (bytes + 255) & ~(size_t)255;
        return p;
    };

    // f16 conversion of all conv weights (d_in index, cout, cin, padded cin)
    struct WD { int i, cout, cin, cinp; };
    const WD wd[21] = {
        {1, 32,  3, 32}, {3, 32, 32, 32}, {5, 64, 32, 32},
        {7, 64,  3, 32}, {9, 64, 64, 64}, {11,128, 64, 64},
        {13,64,  3, 32}, {15,96, 64, 64}, {17,128, 96, 96},
        {19,64, 323,352},{21,64, 64, 64}, {23,128, 64, 64},
        {25,128,323,352},{27,128,128,128},{29,256,128,128},
        {31,128,323,352},{33,128,128,128},{35,256,128,128},
        {37,256,643,672},{39,512,256,256},{41,1024,512,512},
    };
    f16* wp[21];
    for (int i = 0; i < 21; ++i) {
        int tot = wd[i].cout * wd[i].cinp;
        wp[i] = (f16*)alloc((size_t)tot * sizeof(f16));
        convw_kernel<<<(tot + 255) / 256, 256, 0, stream>>>(
            (const float*)d_in[wd[i].i], wp[i], wd[i].cout, wd[i].cin, wd[i].cinp);
    }

    int*   fps1  = (int*)  alloc((size_t)B * 512 * sizeof(int));
    float* l1xyz = (float*)alloc((size_t)B * 512 * 3 * sizeof(float));
    int*   idx10 = (int*)  alloc((size_t)B * 512 * 16  * sizeof(int));
    int*   idx11 = (int*)  alloc((size_t)B * 512 * 32  * sizeof(int));
    int*   idx12 = (int*)  alloc((size_t)B * 512 * 128 * sizeof(int));
    float* l1pts = (float*)alloc((size_t)B * 512 * 320 * sizeof(float));
    int*   fps2  = (int*)  alloc((size_t)B * 128 * sizeof(int));
    float* l2xyz = (float*)alloc((size_t)B * 128 * 3 * sizeof(float));
    int*   idx20 = (int*)  alloc((size_t)B * 128 * 32  * sizeof(int));
    int*   idx21 = (int*)  alloc((size_t)B * 128 * 64  * sizeof(int));
    int*   idx22 = (int*)  alloc((size_t)B * 128 * 128 * sizeof(int));
    float* l2pts = (float*)alloc((size_t)B * 128 * 640 * sizeof(float));
    float* sa3o  = (float*)alloc((size_t)B * 1024 * sizeof(float));
    float* fc1o  = (float*)alloc((size_t)B * 512 * sizeof(float));
    float* fc2o  = (float*)alloc((size_t)B * 256 * sizeof(float));

    const float* bias(const int k);  // fwd decl dummy (unused)

    // ---- SA1 -------------------------------------------------------------
    fps_kernel<<<B, 256, 0, stream>>>(pc, N, 512, fps1, l1xyz);
    {
        int total = B * 512;
        int grid  = (total * 32 + 255) / 256;
        ballquery_kernel<<<grid, 256, 0, stream>>>(pc, N, l1xyz, total, 512, 0.1f * 0.1f, 16,  idx10);
        ballquery_kernel<<<grid, 256, 0, stream>>>(pc, N, l1xyz, total, 512, 0.2f * 0.2f, 32,  idx11);
        ballquery_kernel<<<grid, 256, 0, stream>>>(pc, N, l1xyz, total, 512, 0.4f * 0.4f, 128, idx12);
    }
    sa_mlp_kernel<0, 16, 32, 32, 32, 64, 16, 0><<<B * 512, 256, 0, stream>>>(
        pc, N, nullptr, l1xyz, idx10,
        wp[0], (const float*)d_in[2], wp[1], (const float*)d_in[4], wp[2], (const float*)d_in[6],
        l1pts, 512, 320, 0);
    sa_mlp_kernel<0, 32, 32, 64, 64, 128, 32, 0><<<B * 512, 256, 0, stream>>>(
        pc, N, nullptr, l1xyz, idx11,
        wp[3], (const float*)d_in[8], wp[4], (const float*)d_in[10], wp[5], (const float*)d_in[12],
        l1pts, 512, 320, 64);
    sa_mlp_kernel<0, 128, 32, 64, 96, 128, 32, 0><<<B * 512, 256, 0, stream>>>(
        pc, N, nullptr, l1xyz, idx12,
        wp[6], (const float*)d_in[14], wp[7], (const float*)d_in[16], wp[8], (const float*)d_in[18],
        l1pts, 512, 320, 192);

    // ---- SA2 -------------------------------------------------------------
    fps_kernel<<<B, 256, 0, stream>>>(l1xyz, 512, 128, fps2, l2xyz);
    {
        int total = B * 128;
        int grid  = (total * 32 + 255) / 256;
        ballquery_kernel<<<grid, 256, 0, stream>>>(l1xyz, 512, l2xyz, total, 128, 0.2f * 0.2f, 32,  idx20);
        ballquery_kernel<<<grid, 256, 0, stream>>>(l1xyz, 512, l2xyz, total, 128, 0.4f * 0.4f, 64,  idx21);
        ballquery_kernel<<<grid, 256, 0, stream>>>(l1xyz, 512, l2xyz, total, 128, 0.8f * 0.8f, 128, idx22);
    }
    sa_mlp_kernel<1, 32, 352, 64, 64, 128, 32, 320><<<B * 128, 256, 0, stream>>>(
        l1xyz, 512, l1pts, l2xyz, idx20,
        wp[9], (const float*)d_in[20], wp[10], (const float*)d_in[22], wp[11], (const float*)d_in[24],
        l2pts, 128, 640, 0);
    sa_mlp_kernel<1, 64, 352, 128, 128, 256, 32, 320><<<B * 128, 256, 0, stream>>>(
        l1xyz, 512, l1pts, l2xyz, idx21,
        wp[12], (const float*)d_in[26], wp[13], (const float*)d_in[28], wp[14], (const float*)d_in[30],
        l2pts, 128, 640, 128);
    sa_mlp_kernel<1, 128, 352, 128, 128, 256, 32, 320><<<B * 128, 256, 0, stream>>>(
        l1xyz, 512, l1pts, l2xyz, idx22,
        wp[15], (const float*)d_in[32], wp[16], (const float*)d_in[34], wp[17], (const float*)d_in[36],
        l2pts, 128, 640, 384);

    // ---- SA3 (group_all) -------------------------------------------------
    sa_mlp_kernel<2, 128, 672, 256, 512, 1024, 16, 640><<<B, 256, 0, stream>>>(
        l2xyz, 128, l2pts, nullptr, nullptr,
        wp[18], (const float*)d_in[38], wp[19], (const float*)d_in[40], wp[20], (const float*)d_in[42],
        sa3o, 1, 1024, 0);

    // ---- FC head ---------------------------------------------------------
    fc_kernel<<<(B * 512 + 255) / 256, 256, 0, stream>>>(
        sa3o, 1024, (const float*)d_in[43], (const float*)d_in[44], fc1o, 512, B, 1);
    fc_kernel<<<(B * 256 + 255) / 256, 256, 0, stream>>>(
        fc1o, 512, (const float*)d_in[45], (const float*)d_in[46], fc2o, 256, B, 1);
    fc_kernel<<<(B * 40 + 255) / 256, 256, 0, stream>>>(
        fc2o, 256, (const float*)d_in[47], (const float*)d_in[48], (float*)d_out, 40, B, 0);
}
